// CNN_RNN_88347477278730
// MI455X (gfx1250) — matrix operand order repaired and checked by reading, not compiled, not run on hardware
//
#include <hip/hip_runtime.h>
#include <math.h>

// ---------------- problem constants ----------------
#define Bn 64
#define Tn 512
#define En 768
#define Hn 256
#define G3 768            // 3*H
#define NFp 112           // NF=100 padded to 7*16
#define KSTEP 64          // K-tile per LDS stage (2 wmma k-iters)
#define BSTRIDE 72        // LDS row stride for B tile (64 + 8 pad)

typedef __bf16 bf16;
typedef __attribute__((ext_vector_type(16))) __bf16 v16bf;
typedef __attribute__((ext_vector_type(8)))  __bf16 v8bf;
typedef __attribute__((ext_vector_type(8)))  float  v8f;
typedef __attribute__((address_space(3))) bf16 lds_bf16;

// ---------------- CDNA5 async copy helpers ----------------
// GLOBAL_LOAD_ASYNC_TO_LDS_B128: LDS[vdst] = MEM[vaddr], tracked by ASYNCcnt.
__device__ __forceinline__ void async_copy_b128(lds_bf16* lptr, const bf16* gptr) {
  asm volatile("global_load_async_to_lds_b128 %0, %1, off"
               : : "v"(lptr), "v"(gptr) : "memory");
}
__device__ __forceinline__ void wait_async() {
  asm volatile("s_wait_asynccnt 0x0" : : : "memory");
}

// ---------------- WMMA fragment helpers (wave32, 16x16x32 bf16) ----------------
// A (16x32, MxK): lanes 0-15 -> M rows (khalf=0), lanes 16-31 -> same rows, khalf=1.
__device__ __forceinline__ v16bf load_a_frag(const bf16* A, int lda, int m0, int k0, int mmax) {
  int lane = threadIdx.x & 31;
  int m = m0 + (lane & 15);
  if (m > mmax) m = mmax;                 // row clamp (used by conv windows)
  int kh = lane >> 4;                     // 0 or 1
  const bf16* p = A + (size_t)m * lda + k0 + kh * 8;
  v8bf lo = *(const v8bf*)(p);            // K = k0+kh*8 .. +7
  v8bf hi = *(const v8bf*)(p + 16);       // K = k0+16+kh*8 .. +7
  v16bf r;
#pragma unroll
  for (int i = 0; i < 8; ++i) { r[i] = lo[i]; r[i + 8] = hi[i]; }
  return r;
}

// A fragment sourced from an LDS-resident tile (emits ds_load_b128)
__device__ __forceinline__ v16bf load_a_frag_lds(const bf16* A, int lda, int k0) {
  int lane = threadIdx.x & 31;
  int m = lane & 15;
  int kh = lane >> 4;
  const bf16* p = A + m * lda + k0 + kh * 8;
  v8bf lo = *(const v8bf*)(p);
  v8bf hi = *(const v8bf*)(p + 16);
  v16bf r;
#pragma unroll
  for (int i = 0; i < 8; ++i) { r[i] = lo[i]; r[i + 8] = hi[i]; }
  return r;
}

// B (32x16, KxN) fragment straight from global W[N,K] (B[k][n] = W[n][k])
__device__ __forceinline__ v16bf load_b_frag_g(const bf16* W, int ldw, int n0, int k0) {
  int lane = threadIdx.x & 31;
  int n = n0 + (lane & 15);
  int ks = k0 + (lane >> 4) * 8;
  const bf16* p = W + (size_t)n * ldw + ks;
  v8bf lo = *(const v8bf*)(p);
  v8bf hi = *(const v8bf*)(p + 16);
  v16bf r;
#pragma unroll
  for (int i = 0; i < 8; ++i) { r[i] = lo[i]; r[i + 8] = hi[i]; }
  return r;
}

// B fragment from an LDS tile laid out [nrow][k] with row stride `stride`.
__device__ __forceinline__ v16bf load_b_frag_lds(const bf16* Bt, int stride, int n0, int k0) {
  int lane = threadIdx.x & 31;
  int n = n0 + (lane & 15);
  int ks = k0 + (lane >> 4) * 8;
  const bf16* p = Bt + n * stride + ks;
  v8bf lo = *(const v8bf*)(p);
  v8bf hi = *(const v8bf*)(p + 16);
  v16bf r;
#pragma unroll
  for (int i = 0; i < 8; ++i) { r[i] = lo[i]; r[i + 8] = hi[i]; }
  return r;
}

// D/C (16x16 f32): lane -> n = lane&15 ; rows m0 + 8*(lane>>4) + v, v=0..7.
__device__ __forceinline__ void store_d(float* Out, int ldo, int m0, int n0, v8f c,
                                        const float* bias) {
  int lane = threadIdx.x & 31;
  int n = n0 + (lane & 15);
  int mb = m0 + (lane >> 4) * 8;
  float bv = bias ? bias[n] : 0.f;
#pragma unroll
  for (int v = 0; v < 8; ++v) Out[(size_t)(mb + v) * ldo + n] = c[v] + bv;
}

// ---------------- bf16 WMMA GEMM: Out[M,N] = A[M,K] @ W[N,K]^T + bias ----------------
// block = 128 threads (4 waves), 64 rows x 64 cols per block.
// Weight tile staged to LDS with async b128 copies, double buffered.
__global__ __launch_bounds__(128) void gemm_bf16_kernel(const bf16* __restrict__ A,
                                                        const bf16* __restrict__ W,
                                                        const float* __restrict__ bias,
                                                        float* __restrict__ Out,
                                                        int M, int N, int K) {
  __shared__ __align__(16) bf16 Bl[2][64 * BSTRIDE];   // 2 x 9 KB
  int tid = threadIdx.x;
  int wave = tid >> 5;
  int m0 = blockIdx.x * 64 + wave * 16;
  int n_base = blockIdx.y * 64;

  auto stage = [&](int buf, int k0) {
#pragma unroll
    for (int c = 0; c < 4; ++c) {
      int ci = tid + c * 128;            // 0..511 (512 x 16B chunks)
      int nrow = ci >> 3;                // 8 chunks per row
      int koff = (ci & 7) * 8;
      const bf16* g = W + (size_t)(n_base + nrow) * K + k0 + koff;
      lds_bf16* l = (lds_bf16*)&Bl[buf][nrow * BSTRIDE + koff];
      async_copy_b128(l, g);
    }
  };

  v8f acc[4] = {};
  int cur = 0;
  stage(0, 0);
  for (int k0 = 0; k0 < K; k0 += KSTEP) {
    wait_async();
    __syncthreads();                     // tile[cur] complete in LDS for all waves
    if (k0 + KSTEP < K) stage(cur ^ 1, k0 + KSTEP);
    if (k0 + KSTEP < K) {
      int mm = m0 + (tid & 15); if (mm > M - 1) mm = M - 1;
      __builtin_prefetch(A + (size_t)mm * K + k0 + KSTEP, 0, 3);
    }
    const bf16* Bt = &Bl[cur][0];
#pragma unroll
    for (int j = 0; j < 2; ++j) {
      v16bf a = load_a_frag(A, K, m0, k0 + j * 32, M - 1);
#pragma unroll
      for (int s = 0; s < 4; ++s) {
        v16bf b = load_b_frag_lds(Bt, BSTRIDE, s * 16, j * 32);
        acc[s] = __builtin_amdgcn_wmma_f32_16x16x32_bf16(false, a, false, b,
                                                         (short)0, acc[s], false, false);
      }
    }
    __syncthreads();                     // all waves done reading tile[cur]
    cur ^= 1;
  }
#pragma unroll
  for (int s = 0; s < 4; ++s) store_d(Out, N, m0, n_base + s * 16, acc[s], bias);
}

// ---------------- elementwise / setup kernels ----------------
__global__ void cvt_kernel(const float* __restrict__ src, bf16* __restrict__ dst, size_t n) {
  size_t i = (size_t)blockIdx.x * blockDim.x + threadIdx.x;
  size_t st = (size_t)gridDim.x * blockDim.x;
  for (; i < n; i += st) dst[i] = (bf16)src[i];
}

__global__ void cvt_conv_kernel(const float* __restrict__ w, bf16* __restrict__ dst, int fsH) {
  int n = NFp * fsH;
  for (int i = blockIdx.x * blockDim.x + threadIdx.x; i < n; i += gridDim.x * blockDim.x) {
    int f = i / fsH, k = i - f * fsH;
    dst[i] = (f < 100) ? (bf16)w[f * fsH + k] : (bf16)0.f;
  }
}

__global__ void lens_kernel(const int* __restrict__ mask, int* __restrict__ lens) {
  __shared__ int s[256];
  int b = blockIdx.x, tid = threadIdx.x;
  int a = 0;
  for (int t = tid; t < Tn; t += 256) a += mask[b * Tn + t];
  s[tid] = a; __syncthreads();
  for (int st = 128; st > 0; st >>= 1) { if (tid < st) s[tid] += s[tid + st]; __syncthreads(); }
  if (tid == 0) lens[b] = s[0];
}

__global__ void maxi_kernel(const int* __restrict__ v, int n, int floorv, int* __restrict__ out) {
  __shared__ int s[64];
  int tid = threadIdx.x;
  int m = floorv;
  for (int i = tid; i < n; i += 64) m = max(m, v[i]);
  s[tid] = m; __syncthreads();
  for (int st = 32; st > 0; st >>= 1) { if (tid < st) s[tid] = max(s[tid], s[tid + st]); __syncthreads(); }
  if (tid == 0) *out = s[0];
}

// ---------------- math helpers ----------------
__device__ __forceinline__ float sigmoidf_(float x) { return 1.f / (1.f + __expf(-x)); }

__device__ __forceinline__ float gumbelf_(unsigned t, unsigned b, unsigned k) {
  unsigned x = t * 1664525u ^ (b * 22695477u + 0x9E3779B9u) ^ (k * 747796405u);
  x ^= x >> 16; x *= 0x7feb352du; x ^= x >> 15; x *= 0x846ca68bu; x ^= x >> 16;
  float u = (float)(x >> 8) * (1.0f / 16777216.0f) + 1e-7f;
  return -__logf(-__logf(u));
}

// ---------------- persistent GRU scans ----------------
// Scans are batch-parallel: 4 workgroups x 16 batches, whole hidden state in LDS,
// t-loop inside the kernel (no per-step launches). Per step:
//   gh = h @ Whh^T + bhh   (WMMA: A = h tile from LDS, B = Whh from global/L2)
//   gate update (VALU)     (reads xw stream + gh/h from LDS)
// 16 waves: wave w owns 3 of the 48 N-subtiles; two workgroup barriers per step.

// selector scan: also computes per-batch 2-logit gumbel argmax
__global__ __launch_bounds__(512) void scan_sel_kernel(const float* __restrict__ xw,
                                                       const bf16* __restrict__ Whh,
                                                       const float* __restrict__ bhh,
                                                       const float* __restrict__ Wsel,
                                                       const float* __restrict__ bsel,
                                                       int* __restrict__ selected,
                                                       const int* __restrict__ maxlen) {
  __shared__ __align__(16) float ghs[16 * G3];   // 48 KB
  __shared__ __align__(16) float hfs[16 * Hn];   // 16 KB
  __shared__ __align__(16) bf16  hbs[16 * Hn];   // 8 KB
  int g0 = blockIdx.x * 16;
  int tid = threadIdx.x;
  int wave = tid >> 5;
  int lane = tid & 31;

  for (int i = tid; i < 16 * Hn; i += 512) { hfs[i] = 0.f; hbs[i] = (bf16)0.f; }
  __syncthreads();
  int ml = *maxlen;

  for (int t = 0; t < Tn; ++t) {
    // gh GEMM: 48 subtiles over 16 waves
#pragma unroll
    for (int s = 0; s < 3; ++s) {
      int n0 = (wave * 3 + s) * 16;
      v8f acc = {};
#pragma unroll
      for (int k0 = 0; k0 < Hn; k0 += 32) {
        v16bf a = load_a_frag_lds(hbs, Hn, k0);
        v16bf b = load_b_frag_g(Whh, Hn, n0, k0);
        acc = __builtin_amdgcn_wmma_f32_16x16x32_bf16(false, a, false, b,
                                                      (short)0, acc, false, false);
      }
      int n = n0 + (lane & 15);
      int mb = (lane >> 4) * 8;
      float bv = bhh[n];
#pragma unroll
      for (int v = 0; v < 8; ++v) ghs[(mb + v) * G3 + n] = acc[v] + bv;
    }
    __syncthreads();
    // gate update: 16*256 elements over 512 threads
    for (int e = tid; e < 16 * Hn; e += 512) {
      int bl = e >> 8;
      int j = e & 255;
      int b = g0 + bl;
      const float* xr = xw + ((size_t)b * Tn + t) * G3;
      float r = sigmoidf_(xr[j] + ghs[bl * G3 + j]);
      float z = sigmoidf_(xr[j + Hn] + ghs[bl * G3 + j + Hn]);
      float nn = tanhf(xr[j + 2 * Hn] + r * ghs[bl * G3 + j + 2 * Hn]);
      float hnew = (1.f - z) * nn + z * hfs[e];
      hfs[e] = hnew;
      hbs[e] = (bf16)hnew;
    }
    __syncthreads();
    // selection: wave w -> batch g0+w
    if (t >= 1) {
      float s0 = 0.f, s1 = 0.f;
      for (int j = lane; j < Hn; j += 32) {
        float hv = hfs[wave * Hn + j];
        s0 += hv * Wsel[j];
        s1 += hv * Wsel[Hn + j];
      }
#pragma unroll
      for (int d = 16; d > 0; d >>= 1) {
        s0 += __shfl_down(s0, d, 32);
        s1 += __shfl_down(s1, d, 32);
      }
      if (lane == 0) {
        int b = g0 + wave;
        float l0 = s0 + bsel[0] + gumbelf_(t, b, 0);
        float l1 = s1 + bsel[1] + gumbelf_(t, b, 1);
        int k = (l0 >= l1) ? 0 : 1;
        if (t >= ml - 1) k = 0;
        selected[b * Tn + t] = k;
      }
    }
    __syncthreads();
  }
}

// stacked-layer scan: writes masked bf16 output sequence
__global__ __launch_bounds__(512) void scan_layer_kernel(const float* __restrict__ xw,
                                                         const bf16* __restrict__ Whh,
                                                         const float* __restrict__ bhh,
                                                         bf16* __restrict__ obf,
                                                         const int* __restrict__ new_lens) {
  __shared__ __align__(16) float ghs[16 * G3];
  __shared__ __align__(16) float hfs[16 * Hn];
  __shared__ __align__(16) bf16  hbs[16 * Hn];
  __shared__ int nls[16];
  int g0 = blockIdx.x * 16;
  int tid = threadIdx.x;
  int wave = tid >> 5;
  int lane = tid & 31;

  for (int i = tid; i < 16 * Hn; i += 512) { hfs[i] = 0.f; hbs[i] = (bf16)0.f; }
  if (tid < 16) nls[tid] = new_lens[g0 + tid];
  __syncthreads();

  for (int t = 0; t < Tn; ++t) {
#pragma unroll
    for (int s = 0; s < 3; ++s) {
      int n0 = (wave * 3 + s) * 16;
      v8f acc = {};
#pragma unroll
      for (int k0 = 0; k0 < Hn; k0 += 32) {
        v16bf a = load_a_frag_lds(hbs, Hn, k0);
        v16bf b = load_b_frag_g(Whh, Hn, n0, k0);
        acc = __builtin_amdgcn_wmma_f32_16x16x32_bf16(false, a, false, b,
                                                      (short)0, acc, false, false);
      }
      int n = n0 + (lane & 15);
      int mb = (lane >> 4) * 8;
      float bv = bhh[n];
#pragma unroll
      for (int v = 0; v < 8; ++v) ghs[(mb + v) * G3 + n] = acc[v] + bv;
    }
    __syncthreads();
    for (int e = tid; e < 16 * Hn; e += 512) {
      int bl = e >> 8;
      int j = e & 255;
      int b = g0 + bl;
      const float* xr = xw + ((size_t)b * Tn + t) * G3;
      float r = sigmoidf_(xr[j] + ghs[bl * G3 + j]);
      float z = sigmoidf_(xr[j + Hn] + ghs[bl * G3 + j + Hn]);
      float nn = tanhf(xr[j + 2 * Hn] + r * ghs[bl * G3 + j + 2 * Hn]);
      float hnew = (1.f - z) * nn + z * hfs[e];
      hfs[e] = hnew;
      hbs[e] = (bf16)hnew;
      float valid = (t < nls[bl]) ? 1.f : 0.f;
      obf[((size_t)b * Tn + t) * Hn + j] = (bf16)(hnew * valid);
    }
    __syncthreads();
  }
}

// ---------------- compaction (one block per batch row) ----------------
__global__ __launch_bounds__(256) void compact_kernel(const float* __restrict__ emb,
                                                      const int* __restrict__ selected,
                                                      const int* __restrict__ lens,
                                                      bf16* __restrict__ new_emb,
                                                      int* __restrict__ new_lens) {
  __shared__ int nm[Tn];
  __shared__ int dest[Tn];
  int b = blockIdx.x, tid = threadIdx.x;
  int Lb = lens[b];
  for (int t = tid; t < Tn; t += 256) {
    int sel = (t == 0) ? 1 : selected[b * Tn + t];
    int sv = (t < Lb - 1) ? sel : 0;
    nm[t] = (t == Lb - 1) ? 1 : sv;
  }
  __syncthreads();
  if (tid == 0) {
    int c = 0;
    for (int t = 0; t < Tn; ++t) { if (nm[t]) { dest[t] = c; ++c; } else dest[t] = -1; }
    new_lens[b] = c;
  }
  __syncthreads();
  size_t base = (size_t)b * Tn * En;
  for (int i = tid; i < Tn * En; i += 256) new_emb[base + i] = (bf16)0.f;
  __syncthreads();
  for (int t = 0; t < Tn; ++t) {
    int d = dest[t];
    if (d >= 0) {
      const float* src = emb + base + (size_t)t * En;
      bf16* dst = new_emb + base + (size_t)d * En;
      for (int i = tid; i < En; i += 256) dst[i] = (bf16)src[i];
    }
  }
}

// ---------------- conv (WMMA GEMM over sliding windows) + masked max-pool ----------------
__global__ __launch_bounds__(32) void conv_pool_kernel(const bf16* __restrict__ X,
                                                       const bf16* __restrict__ Wc,
                                                       const float* __restrict__ cb,
                                                       float* __restrict__ pooled,
                                                       const int* __restrict__ ldyn, int fs) {
  __shared__ __align__(16) bf16 Wl[16 * 5 * Hn];   // up to 40 KB (fs=5)
  int b = blockIdx.x, n0 = blockIdx.y * 16;
  const bf16* xb = X + (size_t)b * Tn * Hn;
  int Ld = *ldyn;
  int Pvalid = Ld - fs + 1;
  int Pall = Tn - fs + 1;
  int Kc = fs * Hn;
  int lane = threadIdx.x;

  int nch = (16 * Kc) >> 3;
  int cpr = Kc >> 3;
  for (int ci = lane; ci < nch; ci += 32) {
    int f = ci / cpr;
    int koff = (ci - f * cpr) * 8;
    async_copy_b128((lds_bf16*)&Wl[f * Kc + koff],
                    Wc + (size_t)(n0 + f) * Kc + koff);
  }
  wait_async();
  __syncthreads();

  int ncol = n0 + (lane & 15);
  int mh = lane >> 4;
  float mx = -1e30f;
  float bv = (ncol < 100) ? cb[ncol] : 0.f;
  for (int p0 = 0; p0 < Pall; p0 += 16) {
    v8f acc = {};
    for (int k0 = 0; k0 < Kc; k0 += 32) {
      v16bf a = load_a_frag(xb, Hn, p0, k0, Pall - 1);
      v16bf bb = load_b_frag_lds(Wl, Kc, 0, k0);
      acc = __builtin_amdgcn_wmma_f32_16x16x32_bf16(false, a, false, bb,
                                                    (short)0, acc, false, false);
    }
#pragma unroll
    for (int v = 0; v < 8; ++v) {
      int p = p0 + mh * 8 + v;
      float val = acc[v] + bv;
      val = val > 0.f ? val : 0.f;
      if (p < Pvalid && val > mx) mx = val;
    }
  }
  float other = __shfl_xor(mx, 16, 32);
  mx = fmaxf(mx, other);
  if (lane < 16) pooled[b * NFp + ncol] = mx;
}

// ---------------- final linear head ----------------
__global__ void final_kernel(const float* __restrict__ pooled, const float* __restrict__ Wout,
                             const float* __restrict__ bout, float* __restrict__ out) {
  int b = threadIdx.x;
  if (b >= Bn) return;
  float s = bout[0];
  for (int ci = 0; ci < 3; ++ci)
    for (int f = 0; f < 100; ++f)
      s += pooled[ci * Bn * NFp + b * NFp + f] * Wout[ci * 100 + f];
  out[b] = s;
}

// ---------------- host orchestration ----------------
extern "C" void kernel_launch(void* const* d_in, const int* in_sizes, int n_in,
                              void* d_out, int out_size, void* d_ws, size_t ws_size,
                              hipStream_t stream) {
  (void)in_sizes; (void)n_in; (void)out_size; (void)ws_size;
  const float* emb    = (const float*)d_in[0];
  const int*   mask   = (const int*)d_in[1];
  const float* Wih_c  = (const float*)d_in[2];
  const float* Whh_c  = (const float*)d_in[3];
  const float* bih_c  = (const float*)d_in[4];
  const float* bhh_c  = (const float*)d_in[5];
  const float* Wsel   = (const float*)d_in[6];
  const float* bsel   = (const float*)d_in[7];
  const float* Wih0   = (const float*)d_in[8];
  const float* Whh0   = (const float*)d_in[9];
  const float* bih0   = (const float*)d_in[10];
  const float* bhh0   = (const float*)d_in[11];
  const float* Wih1   = (const float*)d_in[12];
  const float* Whh1   = (const float*)d_in[13];
  const float* bih1   = (const float*)d_in[14];
  const float* bhh1   = (const float*)d_in[15];
  const float* cw3    = (const float*)d_in[16];
  const float* cb3    = (const float*)d_in[17];
  const float* cw4    = (const float*)d_in[18];
  const float* cb4    = (const float*)d_in[19];
  const float* cw5    = (const float*)d_in[20];
  const float* cb5    = (const float*)d_in[21];
  const float* Wout   = (const float*)d_in[22];
  const float* bout   = (const float*)d_in[23];
  float* out = (float*)d_out;

  char* ws = (char*)d_ws;
  size_t off = 0;
  auto alloc = [&](size_t bytes) -> void* {
    void* p = ws + off;
    off = (off + bytes + 255) & ~(size_t)255;
    return p;
  };
  const size_t MT = (size_t)Bn * Tn;                 // 32768
  bf16* emb_bf   = (bf16*)alloc(MT * En * 2);        // 50.3 MB
  bf16* wihc_bf  = (bf16*)alloc((size_t)G3 * En * 2);
  bf16* whhc_bf  = (bf16*)alloc((size_t)G3 * Hn * 2);
  bf16* wih0_bf  = (bf16*)alloc((size_t)G3 * En * 2);
  bf16* whh0_bf  = (bf16*)alloc((size_t)G3 * Hn * 2);
  bf16* wih1_bf  = (bf16*)alloc((size_t)G3 * Hn * 2);
  bf16* whh1_bf  = (bf16*)alloc((size_t)G3 * Hn * 2);
  bf16* wc3_bf   = (bf16*)alloc((size_t)NFp * 3 * Hn * 2);
  bf16* wc4_bf   = (bf16*)alloc((size_t)NFp * 4 * Hn * 2);
  bf16* wc5_bf   = (bf16*)alloc((size_t)NFp * 5 * Hn * 2);
  float* xw      = (float*)alloc(MT * G3 * 4);       // 100.7 MB (reused for all 3 projections)
  bf16* newemb_bf= (bf16*)alloc(MT * En * 2);        // 50.3 MB
  bf16* o1_bf    = (bf16*)alloc(MT * Hn * 2);        // 16.8 MB
  bf16* o2_bf    = (bf16*)alloc(MT * Hn * 2);        // 16.8 MB
  int*   selected= (int*)alloc((size_t)Bn * Tn * 4);
  int*   lens    = (int*)alloc(Bn * 4);
  int*   new_lens= (int*)alloc(Bn * 4);
  int*   maxlen  = (int*)alloc(16);
  int*   ldyn    = (int*)alloc(16);
  float* pooled  = (float*)alloc((size_t)3 * Bn * NFp * 4);

  // ---- conversions to bf16 ----
  cvt_kernel<<<2048, 256, 0, stream>>>(emb, emb_bf, MT * En);
  cvt_kernel<<<256, 256, 0, stream>>>(Wih_c, wihc_bf, (size_t)G3 * En);
  cvt_kernel<<<256, 256, 0, stream>>>(Whh_c, whhc_bf, (size_t)G3 * Hn);
  cvt_kernel<<<256, 256, 0, stream>>>(Wih0, wih0_bf, (size_t)G3 * En);
  cvt_kernel<<<256, 256, 0, stream>>>(Whh0, whh0_bf, (size_t)G3 * Hn);
  cvt_kernel<<<256, 256, 0, stream>>>(Wih1, wih1_bf, (size_t)G3 * Hn);
  cvt_kernel<<<256, 256, 0, stream>>>(Whh1, whh1_bf, (size_t)G3 * Hn);
  cvt_conv_kernel<<<128, 256, 0, stream>>>(cw3, wc3_bf, 3 * Hn);
  cvt_conv_kernel<<<128, 256, 0, stream>>>(cw4, wc4_bf, 4 * Hn);
  cvt_conv_kernel<<<128, 256, 0, stream>>>(cw5, wc5_bf, 5 * Hn);

  // ---- lengths + maxlen ----
  lens_kernel<<<Bn, 256, 0, stream>>>(mask, lens);
  maxi_kernel<<<1, 64, 0, stream>>>(lens, Bn, 0, maxlen);

  // ---- selector GRU: xw = emb @ Wih_c^T + bih_c, then persistent 512-step scan ----
  gemm_bf16_kernel<<<dim3(512, 12), 128, 0, stream>>>(emb_bf, wihc_bf, bih_c, xw,
                                                      (int)MT, G3, En);
  scan_sel_kernel<<<4, 512, 0, stream>>>(xw, whhc_bf, bhh_c, Wsel, bsel, selected, maxlen);

  // ---- compaction ----
  compact_kernel<<<Bn, 256, 0, stream>>>(emb, selected, lens, newemb_bf, new_lens);
  maxi_kernel<<<1, 64, 0, stream>>>(new_lens, Bn, 7, ldyn);

  // ---- GRU layer 0 ----
  gemm_bf16_kernel<<<dim3(512, 12), 128, 0, stream>>>(newemb_bf, wih0_bf, bih0, xw,
                                                      (int)MT, G3, En);
  scan_layer_kernel<<<4, 512, 0, stream>>>(xw, whh0_bf, bhh0, o1_bf, new_lens);

  // ---- GRU layer 1 ----
  gemm_bf16_kernel<<<dim3(512, 12), 128, 0, stream>>>(o1_bf, wih1_bf, bih1, xw,
                                                      (int)MT, G3, Hn);
  scan_layer_kernel<<<4, 512, 0, stream>>>(xw, whh1_bf, bhh1, o2_bf, new_lens);

  // ---- convs + masked max-pool ----
  conv_pool_kernel<<<dim3(Bn, 7), 32, 0, stream>>>(o2_bf, wc3_bf, cb3, pooled + 0 * Bn * NFp, ldyn, 3);
  conv_pool_kernel<<<dim3(Bn, 7), 32, 0, stream>>>(o2_bf, wc4_bf, cb4, pooled + 1 * Bn * NFp, ldyn, 4);
  conv_pool_kernel<<<dim3(Bn, 7), 32, 0, stream>>>(o2_bf, wc5_bf, cb5, pooled + 2 * Bn * NFp, ldyn, 5);

  // ---- final linear ----
  final_kernel<<<1, 64, 0, stream>>>(pooled, Wout, bout, out);
}
